// criterionHinge_62654982914640
// MI455X (gfx1250) — matrix operand order repaired and checked by reading, not compiled, not run on hardware
//
#include <hip/hip_runtime.h>
#include <hip/hip_bf16.h>

// CDNA5 (gfx1250) vector types for WMMA
typedef __attribute__((ext_vector_type(16))) __bf16 v16bf;
typedef __attribute__((ext_vector_type(8)))  __bf16 v8bf;
typedef __attribute__((ext_vector_type(8)))  float  v8f;

// 16-byte int vector matching the async-LDS builtin's parameter type
typedef int v4i_vec __attribute__((vector_size(16)));
typedef __attribute__((address_space(1))) v4i_vec* glb_v4i_ptr;
typedef __attribute__((address_space(3))) v4i_vec* lds_v4i_ptr;

#define D_FEAT 128
#define NTIMES 32

#if defined(__HIP_DEVICE_COMPILE__) && __has_builtin(__builtin_amdgcn_global_load_async_to_lds_b128)
#define USE_ASYNC_LDS 1
#endif

__device__ __forceinline__ unsigned short f2bf_rne(float f) {
    unsigned int u = __float_as_uint(f);
    unsigned int r = u + 0x7FFFu + ((u >> 16) & 1u);
    return (unsigned short)(r >> 16);
}

// Kernel 1: row-normalize X (f32) and store as bf16 into workspace.
// One wave32 per row: each lane handles 4 consecutive features (float4).
__global__ void xnorm_bf16_kernel(const float* __restrict__ X,
                                  unsigned short* __restrict__ XnB, int N) {
    int wave = threadIdx.x >> 5;
    int lane = threadIdx.x & 31;
    int row  = blockIdx.x * 8 + wave;
    if (row >= N) return;
    const float4* xr = (const float4*)(X + (size_t)row * D_FEAT);
    float4 v = xr[lane];
    float ss = v.x * v.x + v.y * v.y + v.z * v.z + v.w * v.w;
#pragma unroll
    for (int o = 16; o > 0; o >>= 1) ss += __shfl_xor(ss, o, 32);
    float inv = rsqrtf(ss);
    ushort4 o4;
    o4.x = f2bf_rne(v.x * inv);
    o4.y = f2bf_rne(v.y * inv);
    o4.z = f2bf_rne(v.z * inv);
    o4.w = f2bf_rne(v.w * inv);
    ((ushort4*)(XnB + (size_t)row * D_FEAT))[lane] = o4;
}

// Kernel 2: one 16x16 pair tile per wave. Async-DMA the prediction j-tile into
// LDS, overlap it with the bf16 WMMA Gram tile (K=128, 4 back-to-back WMMAs
// on preloaded registers), then run the hinge accumulation over 32 t.
__global__ void __launch_bounds__(256)
pair_hinge_kernel(const unsigned short* __restrict__ XnB,
                  const float* __restrict__ pred,
                  const float* __restrict__ scale_p,
                  float* __restrict__ partials,
                  int N, int numTiles) {
    __shared__ float pjs[8][16 * NTIMES];   // per-wave 16x32 prediction tile (16 KB)
    __shared__ float wsum[8];

    const int wave = threadIdx.x >> 5;
    const int lane = threadIdx.x & 31;
    const int m    = lane & 15;
    const int hi   = lane >> 4;

    int tile = blockIdx.x * 8 + wave;
    const bool active = tile < numTiles;
    if (!active) tile = 0;                  // redundant compute, masked out later

    // Map linear tile index -> (tj, tk), tj <= tk, row-major over upper triangle.
    const int TT = N >> 4;
    int tj = 0, rem = tile;
    while (rem >= TT - tj) { rem -= TT - tj; ++tj; }
    const int tk = tj + rem;
    const int jb = tj << 4, kb = tk << 4;

    const float scale = *scale_p;

    // ---- Stage 16x32 j-row prediction tile into LDS (issue DMA first so it
    // overlaps the WMMA Gram computation). Lane (m,hi) owns a 64B half-row. ----
    {
        const float* src = pred + (size_t)(jb + m) * NTIMES + hi * 16;
        float*       dst = &pjs[wave][m * NTIMES + hi * 16];
#ifdef USE_ASYNC_LDS
        glb_v4i_ptr g = (glb_v4i_ptr)(unsigned long long)src;
        lds_v4i_ptr l = (lds_v4i_ptr)(unsigned long long)
                        (unsigned)(unsigned long long)dst;  // low 32 bits = LDS offset
        __builtin_amdgcn_global_load_async_to_lds_b128(g, l, 0,  0);
        __builtin_amdgcn_global_load_async_to_lds_b128(g, l, 16, 0);
        __builtin_amdgcn_global_load_async_to_lds_b128(g, l, 32, 0);
        __builtin_amdgcn_global_load_async_to_lds_b128(g, l, 48, 0);
#else
        const float4* s4 = (const float4*)src;
        float4 t0 = s4[0], t1 = s4[1], t2 = s4[2], t3 = s4[3];
        float4* d4 = (float4*)dst;
        d4[0] = t0; d4[1] = t1; d4[2] = t2; d4[3] = t3;
#endif
    }

    // Per-lane prediction row for k = kb + m (32 floats in registers).
    float4 pk[8];
    const float4* pk4 = (const float4*)(pred + (size_t)(kb + m) * NTIMES);
#pragma unroll
    for (int i = 0; i < 8; ++i) pk[i] = pk4[i];

    // ---- Gram tile via v_wmma_f32_16x16x32_bf16, K = 128 in 4 chunks ----
    // A (16x32 bf16): lane m holds 16 consecutive K starting at hi*16
    // B (32x16 bf16): lane n=m holds 16 consecutive K starting at hi*16
    const __bf16* Arow = (const __bf16*)(XnB) + (size_t)(jb + m) * D_FEAT;
    const __bf16* Brow = (const __bf16*)(XnB) + (size_t)(kb + m) * D_FEAT;
    __builtin_prefetch(Arow, 0, 0);
    __builtin_prefetch(Brow, 0, 0);

    v8bf  aL[4], aH[4];
    v16bf bv[4];
#pragma unroll
    for (int q = 0; q < 4; ++q) {           // preload all K chunks, then WMMA
        aL[q] = *(const v8bf*)(Arow + q * 32 + hi * 16);
        aH[q] = *(const v8bf*)(Arow + q * 32 + hi * 16 + 8);
        bv[q] = *(const v16bf*)(Brow + q * 32 + hi * 16);
    }
    v8f c = {};
#pragma unroll
    for (int q = 0; q < 4; ++q) {
        v16bf a = __builtin_shufflevector(aL[q], aH[q], 0, 1, 2, 3, 4, 5, 6, 7,
                                          8, 9, 10, 11, 12, 13, 14, 15);
        c = __builtin_amdgcn_wmma_f32_16x16x32_bf16(
                /*neg_a=*/false, a, /*neg_b=*/false, bv[q],
                /*c_mod=*/(short)0, c, /*reuse_a=*/false, /*reuse_b=*/false);
    }

    // C/D layout: VGPR r -> row M = r + 8*hi, column N = lane%16.
    float sd[8];
#pragma unroll
    for (int r = 0; r < 8; ++r) sd[r] = scale * (1.0f - c[r]);

    // Ensure the async LDS tile landed, then sync the block.
#ifdef USE_ASYNC_LDS
#if __has_builtin(__builtin_amdgcn_s_wait_asynccnt)
    __builtin_amdgcn_s_wait_asynccnt(0);
#else
    asm volatile("s_wait_asynccnt 0x0" ::: "memory");
#endif
#endif
    __syncthreads();

    float acc = 0.0f;
    const int krow = kb + m;
#pragma unroll
    for (int r = 0; r < 8; ++r) {
        const int jrow = jb + r + (hi << 3);
        const float4* pj4 = (const float4*)(&pjs[wave][(r + (hi << 3)) * NTIMES]);
        const float s_r = sd[r];
        float s = 0.0f;
#pragma unroll
        for (int i = 0; i < 8; ++i) {
            float4 a4 = pj4[i];
            float4 b4 = pk[i];
            s += fmaxf(0.0f, fabsf(a4.x - b4.x) - s_r);
            s += fmaxf(0.0f, fabsf(a4.y - b4.y) - s_r);
            s += fmaxf(0.0f, fabsf(a4.z - b4.z) - s_r);
            s += fmaxf(0.0f, fabsf(a4.w - b4.w) - s_r);
        }
        if (jrow < krow) acc += s;   // only pairs j < k
    }
    if (!active) acc = 0.0f;

    // Deterministic reduction: wave shuffle -> LDS -> one partial per block.
#pragma unroll
    for (int o = 16; o > 0; o >>= 1) acc += __shfl_xor(acc, o, 32);
    if (lane == 0) wsum[wave] = acc;
    __syncthreads();
    if (threadIdx.x == 0) {
        float sblk = 0.0f;
#pragma unroll
        for (int w = 0; w < 8; ++w) sblk += wsum[w];
        partials[blockIdx.x] = sblk;
    }
}

// Kernel 3: sum per-block partials, apply normalization + outer hinge.
__global__ void finalize_kernel(const float* __restrict__ partials, int nPart,
                                const float* __restrict__ target_p,
                                const int* __restrict__ ntimes_p,
                                int N, float* __restrict__ out) {
    __shared__ float buf[256];
    float s = 0.0f;
    for (int i = threadIdx.x; i < nPart; i += 256) s += partials[i];
    buf[threadIdx.x] = s;
    __syncthreads();
    for (int st = 128; st > 0; st >>= 1) {
        if (threadIdx.x < st) buf[threadIdx.x] += buf[threadIdx.x + st];
        __syncthreads();
    }
    if (threadIdx.x == 0) {
        double R = (double)buf[0];
        double denom = (double)N * (double)(N - 1) * (double)(*ntimes_p);
        float mf = (float)(2.0 * R / denom);
        out[0] = fmaxf(0.0f, mf - *target_p);
    }
}

extern "C" void kernel_launch(void* const* d_in, const int* in_sizes, int n_in,
                              void* d_out, int out_size, void* d_ws, size_t ws_size,
                              hipStream_t stream) {
    const float* target = (const float*)d_in[0];          // scalar f32
    const float* pred   = (const float*)d_in[1];          // [N, 32] f32
    const float* X      = (const float*)d_in[2];          // [N, 128] f32
    const int*   ntimes = (const int*)d_in[3];            // scalar int
    const float* scale  = (const float*)d_in[4];          // scalar f32

    const int N = in_sizes[2] / D_FEAT;                   // 2048

    // Workspace layout: [bf16 Xn: N*128*2 bytes] [partials: nBlocks f32]
    unsigned short* XnB = (unsigned short*)d_ws;
    size_t xn_bytes = ((size_t)N * D_FEAT * sizeof(unsigned short) + 255) & ~(size_t)255;
    float* partials = (float*)((char*)d_ws + xn_bytes);

    const int TT = N >> 4;
    const int numTiles = TT * (TT + 1) / 2;               // 8256 for N=2048
    const int nBlocks = (numTiles + 7) / 8;               // 8 waves (tiles) per block

    xnorm_bf16_kernel<<<(N + 7) / 8, 256, 0, stream>>>(X, XnB, N);
    pair_hinge_kernel<<<nBlocks, 256, 0, stream>>>(XnB, pred, scale, partials,
                                                   N, numTiles);
    finalize_kernel<<<1, 256, 0, stream>>>(partials, nBlocks, target, ntimes,
                                           N, (float*)d_out);
}
